// GFASTKAN_Nodes_2173253452199
// MI455X (gfx1250) — matrix operand order repaired and checked by reading, not compiled, not run on hardware
//
#include <hip/hip_runtime.h>

#define N_NODES 50000
#define N_EDGES 500000LL

typedef __attribute__((ext_vector_type(16))) __bf16 bf16x16;
typedef __attribute__((ext_vector_type(8)))  float  f32x8;

union Frag { bf16x16 v; uint4 q[2]; };

__device__ __forceinline__ unsigned short f2bf(float x) {
  unsigned u = __float_as_uint(x);
  u += 0x7FFFu + ((u >> 16) & 1u);
  return (unsigned short)(u >> 16);
}

__global__ void zero_f32(float* __restrict__ p, long long n) {
  long long i = (long long)blockIdx.x * blockDim.x + threadIdx.x;
  if (i < n) p[i] = 0.f;
}

__global__ void copy_to_hcat(const float* __restrict__ x, float* __restrict__ hcat) {
  long long i = (long long)blockIdx.x * blockDim.x + threadIdx.x;
  if (i >= (long long)N_NODES * 128) return;
  long long r = i >> 7;
  int c = (int)(i & 127);
  hcat[r * 384 + c] = x[i];
}

// Re-pack B = [sp_w ; bw] (K x Ncols fp32) into bf16 WMMA fragment order:
// tile (kt,nt) holds 512 bf16; element (lane,e) maps to k = kt*32 + (lane>=16)*16 + e,
// n = nt*16 + (lane&15). Out cols padded with zeros beyond NCreal.
__global__ void prep_weights(const float* __restrict__ spw, const float* __restrict__ bw,
                             unsigned short* __restrict__ dst,
                             int Kspline, int Kbase, int NT, int NCreal) {
  int KT = (Kspline + Kbase) >> 5;
  long long total = (long long)KT * NT * 512;
  long long i = (long long)blockIdx.x * blockDim.x + threadIdx.x;
  if (i >= total) return;
  int e = (int)(i & 15);
  int l = (int)((i >> 4) & 31);
  long long tile = i >> 9;
  int nt = (int)(tile % NT);
  int kt = (int)(tile / NT);
  int n = nt * 16 + (l & 15);
  int k = kt * 32 + ((l >> 4) << 4) + e;
  float v = 0.f;
  if (n < NCreal) {
    v = (k < Kspline) ? spw[(long long)k * NCreal + n]
                      : bw[(long long)(k - Kspline) * NCreal + n];
  }
  dst[i] = f2bf(v);
}

// agg[dst] += relu(x[src] + w) ; float4 per thread, F/4 threads per edge.
__global__ void edge_agg(const float* __restrict__ xin, int xs, int F,
                         const long long* __restrict__ ei,
                         const float* __restrict__ ew,
                         float* __restrict__ agg) {
  int cpe = F >> 2;
  long long tid = (long long)blockIdx.x * blockDim.x + threadIdx.x;
  long long e = tid / cpe;
  if (e >= N_EDGES) return;
  int c = (int)(tid % cpe) << 2;
  long long s = ei[e];
  long long d = ei[N_EDGES + e];
  float w = ew[e];
  float4 v = *(const float4*)(xin + s * (long long)xs + c);
  float* ap = agg + d * (long long)F + c;
  atomicAdd(ap + 0, fmaxf(v.x + w, 0.f));
  atomicAdd(ap + 1, fmaxf(v.y + w, 0.f));
  atomicAdd(ap + 2, fmaxf(v.z + w, 0.f));
  atomicAdd(ap + 3, fmaxf(v.w + w, 0.f));
}

// Fused: y = x + agg ; LayerNorm ; A=[RBF basis | silu(y)] -> LDS bf16 ;
// WMMA bf16 GEMM with double-buffered B fragments ; + (sp_b + bb) ; store.
// MT rows/block; waves = (MT/16)*WS; wave (rg,cg) owns 16 rows x (NTtot/WS) col-tiles.
template<int F, int MT, int NTtot, int WS>
__global__ __launch_bounds__((MT / 16) * WS * 32)
void fastkan_fused(const float* __restrict__ xin, int xs,
                   const float* __restrict__ agg,
                   const float* __restrict__ lnw, const float* __restrict__ lnb,
                   const unsigned short* __restrict__ wfrag,
                   const float* __restrict__ spb, const float* __restrict__ bb,
                   float* __restrict__ out, int os, int ocols, int nrows) {
  constexpr int K   = 5 * F;       // F*G basis + F base
  constexpr int KS  = K + 8;       // padded row stride (keeps 16B align, spreads banks)
  constexpr int KT  = K / 32;      // even for both instantiations
  constexpr int NTW = NTtot / WS;  // col-tiles per wave
  constexpr int RPW = 16 / WS;     // rows per wave in phase 1
  extern __shared__ unsigned short smem[];   // MT * KS bf16

  const int lane = threadIdx.x & 31;
  const int wave = threadIdx.x >> 5;
  const int rg = wave % (MT / 16);           // row-tile group
  const int cg = wave / (MT / 16);           // col group
  const long long rowbase = (long long)blockIdx.x * MT;

  // ---- Phase 1: build A panel in LDS ----
  for (int r = 0; r < RPW; ++r) {
    int row = wave * RPW + r;
    long long grow = rowbase + row;
    float y[F / 32];
    float s = 0.f, sq = 0.f;
    if (grow < nrows) {
#pragma unroll
      for (int i = 0; i < F / 32; ++i) {
        int f = lane + 32 * i;
        float v = xin[grow * (long long)xs + f] + agg[grow * (long long)F + f];
        y[i] = v; s += v; sq += v * v;
      }
    } else {
#pragma unroll
      for (int i = 0; i < F / 32; ++i) y[i] = 0.f;
    }
#pragma unroll
    for (int off = 16; off; off >>= 1) {
      s  += __shfl_xor(s,  off, 32);
      sq += __shfl_xor(sq, off, 32);
    }
    float mu   = s * (1.f / F);
    float var  = sq * (1.f / F) - mu * mu;
    float rstd = rsqrtf(var + 1e-5f);
    unsigned short* arow = smem + row * KS;
#pragma unroll
    for (int i = 0; i < F / 32; ++i) {
      int f = lane + 32 * i;
      float xv = y[i];
      float xn = (xv - mu) * rstd * lnw[f] + lnb[f];
#pragma unroll
      for (int g = 0; g < 4; ++g) {
        float gv = -2.f + (float)g * (4.f / 3.f);   // linspace(-2,2,4)
        float t  = (xn - gv) * 0.75f;               // / denom (=4/3)
        arow[f * 4 + g] = f2bf(__expf(-t * t));
      }
      arow[4 * F + f] = f2bf(xv / (1.f + __expf(-xv)));   // silu base
    }
  }
  __syncthreads();

  // ---- Phase 2: WMMA GEMM, software-pipelined with explicit double buffer.
  //      sched_barrier(0) pins issue order: prefetch clause for step k+1 goes
  //      out before the WMMA group of step k, so waits stay partial. ----
  f32x8 acc[NTW] = {};
  const int koff = (lane >> 4) * 16;
  const unsigned short* abase = smem + (rg * 16 + (lane & 15)) * KS;
  const unsigned short* bbase = wfrag + lane * 16 + (long long)cg * NTW * 512;

  Frag a0, a1, b0[NTW], b1[NTW];

#define LOAD_A(kt, fr)                                              \
  { const unsigned short* ap = abase + (kt) * 32 + koff;            \
    (fr).q[0] = *(const uint4*)(ap);                                \
    (fr).q[1] = *(const uint4*)(ap + 8); }
#define LOAD_B(kt, arr)                                             \
  _Pragma("unroll")                                                 \
  for (int j = 0; j < NTW; ++j) {                                   \
    const unsigned short* bp = bbase + ((long long)(kt) * NTtot + j) * 512; \
    (arr)[j].q[0] = *(const uint4*)(bp);                            \
    (arr)[j].q[1] = *(const uint4*)(bp + 8); }

  LOAD_A(0, a0);
  LOAD_B(0, b0);
  for (int kt = 0; kt < KT; kt += 2) {
    LOAD_B(kt + 1, b1);
    LOAD_A(kt + 1, a1);
    __builtin_amdgcn_sched_barrier(0);
#pragma unroll
    for (int j = 0; j < NTW; ++j)
      acc[j] = __builtin_amdgcn_wmma_f32_16x16x32_bf16(
          false, a0.v, false, b0[j].v, (short)0, acc[j], false, false);
    __builtin_amdgcn_sched_barrier(0);
    if (kt + 2 < KT) {
      LOAD_B(kt + 2, b0);
      LOAD_A(kt + 2, a0);
    }
    __builtin_amdgcn_sched_barrier(0);
#pragma unroll
    for (int j = 0; j < NTW; ++j)
      acc[j] = __builtin_amdgcn_wmma_f32_16x16x32_bf16(
          false, a1.v, false, b1[j].v, (short)0, acc[j], false, false);
    __builtin_amdgcn_sched_barrier(0);
  }
#undef LOAD_A
#undef LOAD_B

  // ---- Epilogue: D layout row = v + 8*(lane>=16), col = lane&15 ----
  const int rsel = (lane >> 4) * 8;
  const int col0 = lane & 15;
#pragma unroll
  for (int j = 0; j < NTW; ++j) {
    int col = (cg * NTW + j) * 16 + col0;
    if (col >= ocols) continue;
    float bias = spb[col] + bb[col];
#pragma unroll
    for (int v = 0; v < 8; ++v) {
      long long grow = rowbase + rg * 16 + rsel + v;
      if (grow < nrows)
        out[grow * (long long)os + col] = acc[j][v] + bias;
    }
  }
}

extern "C" void kernel_launch(void* const* d_in, const int* in_sizes, int n_in,
                              void* d_out, int out_size, void* d_ws, size_t ws_size,
                              hipStream_t stream) {
  const float*     x    = (const float*)d_in[0];
  const long long* ei   = (const long long*)d_in[1];
  const float*     ew   = (const float*)d_in[2];
  const float*     clnw = (const float*)d_in[3];
  const float*     clnb = (const float*)d_in[4];
  const float*     cspw = (const float*)d_in[5];
  const float*     cspb = (const float*)d_in[6];
  const float*     cbw  = (const float*)d_in[7];
  const float*     cbb  = (const float*)d_in[8];
  const float*     olnw = (const float*)d_in[9];
  const float*     olnb = (const float*)d_in[10];
  const float*     ospw = (const float*)d_in[11];
  const float*     ospb = (const float*)d_in[12];
  const float*     obw  = (const float*)d_in[13];
  const float*     obb  = (const float*)d_in[14];
  float* out = (float*)d_out;

  // workspace: hcat [N,384] f32 | agg [N,384] f32 | bf16 fragment weights
  float* hcat = (float*)d_ws;
  float* agg  = hcat + (size_t)N_NODES * 384;
  unsigned short* wf0 = (unsigned short*)(agg + (size_t)N_NODES * 384);
  unsigned short* wf1 = wf0 + 640 * 128;       // K=640, N=128
  unsigned short* wfo = wf1 + 640 * 128;       // K=1920, N=48(pad)

  // ---- weight re-pack (3 small kernels, run once per launch) ----
  {
    long long t = 20LL * 8 * 512;   // KT*NT*512
    prep_weights<<<(unsigned)((t + 255) / 256), 256, 0, stream>>>(cspw, cbw, wf0, 512, 128, 8, 128);
    prep_weights<<<(unsigned)((t + 255) / 256), 256, 0, stream>>>(cspw + 512 * 128, cbw + 128 * 128, wf1, 512, 128, 8, 128);
    long long to = 60LL * 3 * 512;
    prep_weights<<<(unsigned)((to + 255) / 256), 256, 0, stream>>>(ospw, obw, wfo, 1536, 384, 3, 40);
  }

  copy_to_hcat<<<(N_NODES * 128 + 255) / 256, 256, 0, stream>>>(x, hcat);

  const unsigned nb_conv = (N_NODES + 63) / 64;
  const unsigned nb_out  = (N_NODES + 31) / 32;
  const size_t lds_conv = (size_t)64 * (5 * 128 + 8) * 2;   //  82,944 B
  const size_t lds_out  = (size_t)32 * (5 * 384 + 8) * 2;   // 123,392 B

  // ---- layer 0: GINE on x (hcat cols 0..128) -> hcat cols 128..256 ----
  zero_f32<<<(N_NODES * 128 + 255) / 256, 256, 0, stream>>>(agg, (long long)N_NODES * 128);
  edge_agg<<<(unsigned)((N_EDGES * 32 + 255) / 256), 256, 0, stream>>>(hcat, 384, 128, ei, ew, agg);
  fastkan_fused<128, 64, 8, 2><<<nb_conv, 256, lds_conv, stream>>>(
      hcat, 384, agg, clnw, clnb, wf0, cspb, cbb, hcat + 128, 384, 128, N_NODES);

  // ---- layer 1: GINE on h1 (hcat cols 128..256) -> hcat cols 256..384 ----
  zero_f32<<<(N_NODES * 128 + 255) / 256, 256, 0, stream>>>(agg, (long long)N_NODES * 128);
  edge_agg<<<(unsigned)((N_EDGES * 32 + 255) / 256), 256, 0, stream>>>(hcat + 128, 384, 128, ei, ew, agg);
  fastkan_fused<128, 64, 8, 2><<<nb_conv, 256, lds_conv, stream>>>(
      hcat + 128, 384, agg, clnw + 128, clnb + 128, wf1, cspb + 128, cbb + 128,
      hcat + 256, 384, 128, N_NODES);

  // ---- output layer: GINE on hcat [N,384] -> d_out [N,40] ----
  zero_f32<<<(unsigned)(((long long)N_NODES * 384 + 255) / 256), 256, 0, stream>>>(agg, (long long)N_NODES * 384);
  edge_agg<<<(unsigned)((N_EDGES * 96 + 255) / 256), 256, 0, stream>>>(hcat, 384, 384, ei, ew, agg);
  fastkan_fused<384, 32, 3, 1><<<nb_out, 64, lds_out, stream>>>(
      hcat, 384, agg, olnw, olnb, wfo, ospb, obb, out, 40, 40, N_NODES);
}
